// SA_Block_57389353009855
// MI455X (gfx1250) — matrix operand order repaired and checked by reading, not compiled, not run on hardware
//
#include <hip/hip_runtime.h>

#define BB 8
#define CC 256
#define NN 64
#define LL 4096

typedef __attribute__((ext_vector_type(16))) _Float16 v16h;
typedef __attribute__((ext_vector_type(8)))  _Float16 v8h_t;
typedef __attribute__((ext_vector_type(8)))  float    v8f;

// ---------------------------------------------------------------------------
// WMMA fragment loaders (CDNA5 wave32 16x16x32 f16 layouts, CDNA5 ISA)
// A (16x32, MxK): lane<16 row=lane gets K{8g..8g+7} in v0..3, K{16+8g..} in v4..7
// B (32x16, KxN): lane col=lane&15, K{8g..8g+7} in v0..3, K{16+8g..} in v4..7
// D (16x16 f32):  VGPR r <-> row r+8*(lane>>4), col = lane&15
// ---------------------------------------------------------------------------
static __device__ __forceinline__ v16h load_frag_a(const _Float16* base, int ld) {
  const int lane = threadIdx.x & 31;
  const int row  = lane & 15;
  const int g    = lane >> 4;
  const _Float16* p = base + row * ld + 8 * g;
  v8h_t lo = *(const v8h_t*)(p);        // K = 8g .. 8g+7
  v8h_t hi = *(const v8h_t*)(p + 16);   // K = 16+8g .. 16+8g+7
  return __builtin_shufflevector(lo, hi, 0,1,2,3,4,5,6,7,8,9,10,11,12,13,14,15);
}

static __device__ __forceinline__ v16h load_frag_b(const _Float16* base, int ld) {
  // Source memory is row-major [col][k] (i.e. B^T) with leading dim ld.
  const int lane = threadIdx.x & 31;
  const int col  = lane & 15;
  const int g    = lane >> 4;
  const _Float16* p = base + col * ld + 8 * g;
  v8h_t lo = *(const v8h_t*)(p);
  v8h_t hi = *(const v8h_t*)(p + 16);
  return __builtin_shufflevector(lo, hi, 0,1,2,3,4,5,6,7,8,9,10,11,12,13,14,15);
}

static __device__ __forceinline__ v8f wmma16(v16h a, v16h b, v8f c) {
  return __builtin_amdgcn_wmma_f32_16x16x32_f16(false, a, false, b, (short)0, c,
                                                false, false);
}

// CDNA5 async copy: ASYNCcnt-tracked DMA of 64B (4 x b128) global -> LDS.
// CDNA5 ISA: INST_OFFSET is added to BOTH lds and global address.
static __device__ __forceinline__ void async_copy64B(void* lds_dst,
                                                     const void* gsrc) {
  unsigned           l = (unsigned)(uintptr_t)lds_dst;  // low 32b = LDS offset
  unsigned long long g = (unsigned long long)(uintptr_t)gsrc;
  asm volatile(
      "global_load_async_to_lds_b128 %0, %1, off\n\t"
      "global_load_async_to_lds_b128 %0, %1, off offset:16\n\t"
      "global_load_async_to_lds_b128 %0, %1, off offset:32\n\t"
      "global_load_async_to_lds_b128 %0, %1, off offset:48"
      :: "v"(l), "v"(g) : "memory");
}

static __device__ __forceinline__ void wait_async0() {
  asm volatile("s_wait_asynccnt 0x0" ::: "memory");
}

// ---------------------------------------------------------------------------
// Kernel 0: f32 -> f16 weight conversion (one-time per launch, tiny)
// ---------------------------------------------------------------------------
__global__ void cvt_f32_f16_kernel(const float* __restrict__ src,
                                   _Float16* __restrict__ dst, int n) {
  for (int i = blockIdx.x * blockDim.x + threadIdx.x; i < n;
       i += gridDim.x * blockDim.x)
    dst[i] = (_Float16)src[i];
}

// ---------------------------------------------------------------------------
// Kernel 1: QKV projection.  Q/K/V[b,l,n] = sum_c X[b,c,l] * W[n,c] + bias[n]
// Direct pointer per matrix (no pointer arrays) so weight loads stay GLOBAL.
// ---------------------------------------------------------------------------
static __device__ __forceinline__ void proj_mat(
    const v16h a[8], const _Float16* __restrict__ W,
    const float* __restrict__ bias, _Float16* __restrict__ outp, int b, int l0,
    int wave, int hl, int g) {
  v8f acc[4];
#pragma unroll
  for (int ct = 0; ct < 4; ++ct) {
    float bv = bias[ct * 16 + hl];
#pragma unroll
    for (int r = 0; r < 8; ++r) acc[ct][r] = bv;
  }
#pragma unroll
  for (int kc = 0; kc < 8; ++kc)
#pragma unroll
    for (int ct = 0; ct < 4; ++ct)
      acc[ct] =
          wmma16(a[kc], load_frag_b(W + (ct * 16) * CC + 32 * kc, CC), acc[ct]);
#pragma unroll
  for (int ct = 0; ct < 4; ++ct)
#pragma unroll
    for (int r = 0; r < 8; ++r) {
      int l = l0 + 16 * wave + r + 8 * g;
      outp[((size_t)b * LL + l) * NN + ct * 16 + hl] = (_Float16)acc[ct][r];
    }
}

__global__ __launch_bounds__(128) void proj_qkv_kernel(
    const float* __restrict__ X, const _Float16* __restrict__ Wa,
    const _Float16* __restrict__ Wb, const _Float16* __restrict__ Wg,
    const float* __restrict__ ba, const float* __restrict__ bbias,
    const float* __restrict__ bg, _Float16* __restrict__ Qo,
    _Float16* __restrict__ Ko, _Float16* __restrict__ Vo) {
  const int b    = blockIdx.y;
  const int l0   = blockIdx.x * 64;
  const int tid  = threadIdx.x;
  const int wave = tid >> 5;
  const int lane = tid & 31;
  const int hl   = lane & 15, g = lane >> 4;

  __shared__ _Float16 Xt[64 * 264];  // X^T tile [l][c], stride 264 (pad)

  for (int c = tid; c < CC; c += 128) {
    const float* xr = X + ((size_t)b * CC + c) * LL + l0;
#pragma unroll
    for (int l = 0; l < 64; l += 4) {
      float4 v = *(const float4*)(xr + l);
      Xt[(l + 0) * 264 + c] = (_Float16)v.x;
      Xt[(l + 1) * 264 + c] = (_Float16)v.y;
      Xt[(l + 2) * 264 + c] = (_Float16)v.z;
      Xt[(l + 3) * 264 + c] = (_Float16)v.w;
    }
  }
  __syncthreads();

  // hoist the 8 A fragments (shared by all three projection matrices)
  v16h a[8];
#pragma unroll
  for (int kc = 0; kc < 8; ++kc)
    a[kc] = load_frag_a(Xt + (16 * wave) * 264 + 32 * kc, 264);

  proj_mat(a, Wa, ba, Qo, b, l0, wave, hl, g);
  proj_mat(a, Wb, bbias, Ko, b, l0, wave, hl, g);
  proj_mat(a, Wg, bg, Vo, b, l0, wave, hl, g);
}

// ---------------------------------------------------------------------------
// Kernel 2: flash attention, d = 64, seq = 4096, exact online softmax.
// Double-buffered tiles: K via async global->LDS DMA (ASYNCcnt), V staged
// transposed through registers; prefetch of tile j+1 overlaps WMMAs of tile j.
// ---------------------------------------------------------------------------
__global__ __launch_bounds__(128) void attn_kernel(
    const _Float16* __restrict__ Qm, const _Float16* __restrict__ Km,
    const _Float16* __restrict__ Vm, _Float16* __restrict__ Om) {
  const int b    = blockIdx.y;
  const int i0   = blockIdx.x * 64;
  const int tid  = threadIdx.x;
  const int wave = tid >> 5, lane = tid & 31;
  const int hl   = lane & 15, g = lane >> 4;

  __shared__ _Float16 Klds[2][64 * 72];   // K tile [j][k], double buffered
  __shared__ _Float16 Vtlds[2][64 * 72];  // V^T tile [n][j], double buffered
  __shared__ _Float16 Plds[4][16 * 72];   // per-wave P relayout buffer

  const _Float16* qbase = Qm + ((size_t)b * LL + i0 + 16 * wave) * NN;
  const v16h qa0 = load_frag_a(qbase, NN);       // k = 0..31
  const v16h qa1 = load_frag_a(qbase + 32, NN);  // k = 32..63

  v8f   o[4];
  float mrow[8], lrow[8];
#pragma unroll
  for (int ct = 0; ct < 4; ++ct)
#pragma unroll
    for (int r = 0; r < 8; ++r) o[ct][r] = 0.f;
#pragma unroll
  for (int r = 0; r < 8; ++r) { mrow[r] = -1e30f; lrow[r] = 0.f; }

  const int srow = tid >> 1;
  const int shb  = (tid & 1) * 32;

  // stage K (async DMA) + V (register transpose) for tile at column j0
  auto stage = [&](int j0, int buf) {
    async_copy64B(&Klds[buf][srow * 72 + shb],
                  Km + ((size_t)b * LL + j0 + srow) * NN + shb);
    const _Float16* vsrc = Vm + ((size_t)b * LL + j0 + srow) * NN + shb;
#pragma unroll
    for (int i = 0; i < 32; i += 8) {
      v8h_t v = *(const v8h_t*)(vsrc + i);
#pragma unroll
      for (int e = 0; e < 8; ++e) Vtlds[buf][(shb + i + e) * 72 + srow] = v[e];
    }
  };

  stage(0, 0);  // prefetch first tile

  for (int jt = 0; jt < LL / 64; ++jt) {
    const int cur = jt & 1;
    // own async DMA (of buffer `cur`) done, then make all waves' data visible;
    // this barrier also guarantees prior readers of buffer cur^1 are retired.
    wait_async0();
    __syncthreads();
    if (jt + 1 < LL / 64) stage((jt + 1) * 64, cur ^ 1);  // overlap with WMMAs

    const _Float16* Kc = &Klds[cur][0];
    const _Float16* Vc = &Vtlds[cur][0];

    // ---- S = Q * K^T  (16 x 64 per wave) ----
    v8f S[4];
#pragma unroll
    for (int jtile = 0; jtile < 4; ++jtile) {
      v8f z;
#pragma unroll
      for (int r = 0; r < 8; ++r) z[r] = 0.f;
      z = wmma16(qa0, load_frag_b(Kc + jtile * 16 * 72, 72), z);
      z = wmma16(qa1, load_frag_b(Kc + jtile * 16 * 72 + 32, 72), z);
      S[jtile] = z;
    }

    // ---- online softmax ----
    float mnew[8], corr[8];
#pragma unroll
    for (int r = 0; r < 8; ++r) {
      float tm = fmaxf(fmaxf(S[0][r], S[1][r]), fmaxf(S[2][r], S[3][r]));
      tm = fmaxf(tm, __shfl_xor(tm, 1, 32));
      tm = fmaxf(tm, __shfl_xor(tm, 2, 32));
      tm = fmaxf(tm, __shfl_xor(tm, 4, 32));
      tm = fmaxf(tm, __shfl_xor(tm, 8, 32));
      mnew[r] = fmaxf(mrow[r], tm);
      corr[r] = __expf(mrow[r] - mnew[r]);
      mrow[r] = mnew[r];
    }
#pragma unroll
    for (int jtile = 0; jtile < 4; ++jtile)
#pragma unroll
      for (int r = 0; r < 8; ++r) S[jtile][r] = __expf(S[jtile][r] - mnew[r]);
#pragma unroll
    for (int r = 0; r < 8; ++r) {
      float rs = S[0][r] + S[1][r] + S[2][r] + S[3][r];
      rs += __shfl_xor(rs, 1, 32);
      rs += __shfl_xor(rs, 2, 32);
      rs += __shfl_xor(rs, 4, 32);
      rs += __shfl_xor(rs, 8, 32);
      lrow[r] = lrow[r] * corr[r] + rs;
    }
#pragma unroll
    for (int ct = 0; ct < 4; ++ct)
#pragma unroll
      for (int r = 0; r < 8; ++r) o[ct][r] *= corr[r];

    // ---- relayout P: D-layout -> A-layout via per-wave LDS buffer ----
    _Float16* pb = &Plds[wave][0];
#pragma unroll
    for (int jtile = 0; jtile < 4; ++jtile)
#pragma unroll
      for (int r = 0; r < 8; ++r)
        pb[(r + 8 * g) * 72 + jtile * 16 + hl] = (_Float16)S[jtile][r];

    asm volatile("s_wait_dscnt 0x0" ::: "memory");  // wave-local LDS RAW fence

    // ---- O += P * V ----
#pragma unroll
    for (int kc = 0; kc < 2; ++kc) {
      v16h pa = load_frag_a(pb + 32 * kc, 72);
#pragma unroll
      for (int ct = 0; ct < 4; ++ct) {
        v16h vb = load_frag_b(Vc + ct * 16 * 72 + 32 * kc, 72);
        o[ct] = wmma16(pa, vb, o[ct]);
      }
    }
  }

  // normalize + store O[b,l,n] in f16
#pragma unroll
  for (int ct = 0; ct < 4; ++ct)
#pragma unroll
    for (int r = 0; r < 8; ++r) {
      int i = i0 + 16 * wave + r + 8 * g;
      Om[((size_t)b * LL + i) * NN + ct * 16 + hl] =
          (_Float16)(o[ct][r] / lrow[r]);
    }
}

// ---------------------------------------------------------------------------
// Kernel 3: epilogue  P[b,c,l] = conv_w@O + F_w@X + conv_b + F_b
// ---------------------------------------------------------------------------
__global__ __launch_bounds__(128) void epilogue_kernel(
    const float* __restrict__ X, const _Float16* __restrict__ Om,
    const _Float16* __restrict__ Ch, const float* __restrict__ cb,
    const _Float16* __restrict__ Fh, const float* __restrict__ fb,
    float* __restrict__ Out) {
  const int b    = blockIdx.y;
  const int l0   = blockIdx.x * 64;
  const int tid  = threadIdx.x;
  const int wave = tid >> 5, lane = tid & 31;
  const int hl   = lane & 15, g = lane >> 4;

  __shared__ _Float16 Xt[64 * 264];
  __shared__ float    Obuf[64 * 65];

  for (int c = tid; c < CC; c += 128) {
    const float* xr = X + ((size_t)b * CC + c) * LL + l0;
#pragma unroll
    for (int l = 0; l < 64; l += 4) {
      float4 v = *(const float4*)(xr + l);
      Xt[(l + 0) * 264 + c] = (_Float16)v.x;
      Xt[(l + 1) * 264 + c] = (_Float16)v.y;
      Xt[(l + 2) * 264 + c] = (_Float16)v.z;
      Xt[(l + 3) * 264 + c] = (_Float16)v.w;
    }
  }
  const _Float16* obase = Om + ((size_t)b * LL + l0 + 16 * wave) * NN;
  const v16h oa0 = load_frag_a(obase, NN);
  const v16h oa1 = load_frag_a(obase + 32, NN);
  __syncthreads();

  // hoist X^T A-fragments once; reused by all four c-chunks
  v16h xa[8];
#pragma unroll
  for (int kc = 0; kc < 8; ++kc)
    xa[kc] = load_frag_a(Xt + (16 * wave) * 264 + 32 * kc, 264);

  for (int c0 = 0; c0 < CC; c0 += 64) {
    v8f acc[4];
#pragma unroll
    for (int ct = 0; ct < 4; ++ct) {
      float bv = cb[c0 + ct * 16 + hl] + fb[c0 + ct * 16 + hl];
#pragma unroll
      for (int r = 0; r < 8; ++r) acc[ct][r] = bv;
    }
    // conv path: K = N = 64
#pragma unroll
    for (int ct = 0; ct < 4; ++ct) {
      acc[ct] = wmma16(oa0, load_frag_b(Ch + (c0 + ct * 16) * NN, NN), acc[ct]);
      acc[ct] =
          wmma16(oa1, load_frag_b(Ch + (c0 + ct * 16) * NN + 32, NN), acc[ct]);
    }
    // residual F path: K = C = 256
#pragma unroll
    for (int kc = 0; kc < 8; ++kc)
#pragma unroll
      for (int ct = 0; ct < 4; ++ct)
        acc[ct] =
            wmma16(xa[kc], load_frag_b(Fh + (c0 + ct * 16) * CC + 32 * kc, CC),
                   acc[ct]);
    // transpose through LDS so stores along l are coalesced 128B runs
#pragma unroll
    for (int ct = 0; ct < 4; ++ct)
#pragma unroll
      for (int r = 0; r < 8; ++r)
        Obuf[(16 * wave + r + 8 * g) * 65 + ct * 16 + hl] = acc[ct][r];
    __syncthreads();
    {
      int    cr  = tid >> 1;
      int    lh  = (tid & 1) * 32;
      float* dst = Out + ((size_t)b * CC + c0 + cr) * LL + l0 + lh;
#pragma unroll
      for (int i = 0; i < 32; i += 4) {
        float4 v;
        v.x = Obuf[(lh + i + 0) * 65 + cr];
        v.y = Obuf[(lh + i + 1) * 65 + cr];
        v.z = Obuf[(lh + i + 2) * 65 + cr];
        v.w = Obuf[(lh + i + 3) * 65 + cr];
        *(float4*)(dst + i) = v;
      }
    }
    __syncthreads();
  }
}

// ---------------------------------------------------------------------------
extern "C" void kernel_launch(void* const* d_in, const int* in_sizes, int n_in,
                              void* d_out, int out_size, void* d_ws,
                              size_t ws_size, hipStream_t stream) {
  (void)in_sizes; (void)n_in; (void)out_size; (void)ws_size;
  const float* X      = (const float*)d_in[0];
  const float* Wa_w   = (const float*)d_in[1];
  const float* Wa_b   = (const float*)d_in[2];
  const float* Wb_w   = (const float*)d_in[3];
  const float* Wb_b   = (const float*)d_in[4];
  const float* Wg_w   = (const float*)d_in[5];
  const float* Wg_b   = (const float*)d_in[6];
  const float* conv_w = (const float*)d_in[7];
  const float* conv_b = (const float*)d_in[8];
  const float* F_w    = (const float*)d_in[9];
  const float* F_b    = (const float*)d_in[10];
  float*       Out    = (float*)d_out;

  char*     p    = (char*)d_ws;
  _Float16* Wa_h = (_Float16*)p; p += (size_t)NN * CC * 2;
  _Float16* Wb_h = (_Float16*)p; p += (size_t)NN * CC * 2;
  _Float16* Wg_h = (_Float16*)p; p += (size_t)NN * CC * 2;
  _Float16* Ch   = (_Float16*)p; p += (size_t)CC * NN * 2;
  _Float16* FhW  = (_Float16*)p; p += (size_t)CC * CC * 2;
  _Float16* Qh   = (_Float16*)p; p += (size_t)BB * LL * NN * 2;
  _Float16* Kh   = (_Float16*)p; p += (size_t)BB * LL * NN * 2;
  _Float16* Vh   = (_Float16*)p; p += (size_t)BB * LL * NN * 2;
  _Float16* Oh   = (_Float16*)p; p += (size_t)BB * LL * NN * 2;

  cvt_f32_f16_kernel<<<64, 256, 0, stream>>>(Wa_w, Wa_h, NN * CC);
  cvt_f32_f16_kernel<<<64, 256, 0, stream>>>(Wb_w, Wb_h, NN * CC);
  cvt_f32_f16_kernel<<<64, 256, 0, stream>>>(Wg_w, Wg_h, NN * CC);
  cvt_f32_f16_kernel<<<64, 256, 0, stream>>>(conv_w, Ch, CC * NN);
  cvt_f32_f16_kernel<<<256, 256, 0, stream>>>(F_w, FhW, CC * CC);

  dim3 grid(LL / 64, BB);
  proj_qkv_kernel<<<grid, 128, 0, stream>>>(X, Wa_h, Wb_h, Wg_h, Wa_b, Wb_b,
                                            Wg_b, Qh, Kh, Vh);
  attn_kernel<<<grid, 128, 0, stream>>>(Qh, Kh, Vh, Oh);
  epilogue_kernel<<<grid, 128, 0, stream>>>(X, Oh, Ch, conv_b, FhW, F_b, Out);
}
